// NeRF_9165460209931
// MI455X (gfx1250) — matrix operand order repaired and checked by reading, not compiled, not run on hardware
//
#include <hip/hip_runtime.h>

typedef __attribute__((ext_vector_type(16))) _Float16 v16h;
typedef __attribute__((ext_vector_type(8)))  _Float16 v8h;
typedef __attribute__((ext_vector_type(2)))  _Float16 v2h;
typedef __attribute__((ext_vector_type(8)))  float    v8f;

#define T_HASH 4096
#define LVLS   16

// packed f16 weight layout (offsets in halves), each matrix stored [N][K]
#define OFF_D0 0        // 64 x 32
#define OFF_D1 2048     // 64 x 64
#define OFF_D2 6144     // 16 x 64
#define OFF_C0 7168     // 64 x 32 (K padded 16->32 with zeros)
#define OFF_C1 9216     // 64 x 64
#define OFF_C2 13312    // 64 x 64
#define OFF_C3 17408    // 16 x 64 (N padded 3->16 with zeros)
#define WS_HALVES 18432
#define TAB_HALVES (LVLS * T_HASH * 2)   // 131072 f16 table entries after pack

// ---------------------------------------------------------------------------
// Weight repack: f32 [K][N] (row-major) -> f16 [N][K] (WMMA B-fragment layout)
// ---------------------------------------------------------------------------
__global__ __launch_bounds__(256) void pack_weights(
    const float* __restrict__ dW0, const float* __restrict__ dW1,
    const float* __restrict__ dW2, const float* __restrict__ cW0,
    const float* __restrict__ cW1, const float* __restrict__ cW2,
    const float* __restrict__ cW3, _Float16* __restrict__ ws)
{
    int i = blockIdx.x * 256 + threadIdx.x;
    if (i >= WS_HALVES) return;
    if (i < OFF_D1) {                       // dW0: [32][64] -> [64][32]
        int n = i >> 5, k = i & 31;
        ws[i] = (_Float16)dW0[k * 64 + n];
    } else if (i < OFF_D2) {                // dW1: [64][64] -> [64][64]
        int j = i - OFF_D1, n = j >> 6, k = j & 63;
        ws[i] = (_Float16)dW1[k * 64 + n];
    } else if (i < OFF_C0) {                // dW2: [64][16] -> [16][64]
        int j = i - OFF_D2, n = j >> 6, k = j & 63;
        ws[i] = (_Float16)dW2[k * 16 + n];
    } else if (i < OFF_C1) {                // cW0: [16][64] -> [64][32], K-pad
        int j = i - OFF_C0, n = j >> 5, k = j & 31;
        ws[i] = (_Float16)((k < 16) ? cW0[k * 64 + n] : 0.f);
    } else if (i < OFF_C2) {                // cW1
        int j = i - OFF_C1, n = j >> 6, k = j & 63;
        ws[i] = (_Float16)cW1[k * 64 + n];
    } else if (i < OFF_C3) {                // cW2
        int j = i - OFF_C2, n = j >> 6, k = j & 63;
        ws[i] = (_Float16)cW2[k * 64 + n];
    } else {                                // cW3: [64][3] -> [16][64], N-pad
        int j = i - OFF_C3, n = j >> 6, k = j & 63;
        ws[i] = (_Float16)((n < 3) ? cW3[k * 3 + n] : 0.f);
    }
}

// Hash tables f32 -> f16 (feature pairs stay adjacent: one b32 gather/corner)
__global__ __launch_bounds__(256) void pack_tables(
    const float* __restrict__ t, _Float16* __restrict__ out)
{
    int i = blockIdx.x * 256 + threadIdx.x;
    if (i < TAB_HALVES) out[i] = (_Float16)t[i];
}

// ---------------------------------------------------------------------------
// WMMA fragment loaders (CDNA5 16-bit layouts, wave32)
// ---------------------------------------------------------------------------
// A (16x32, f16): lanes 0-15 = rows M=0..15 holding K chunks [0..7],[16..23];
//                 lanes 16-31 = same rows holding K chunks [8..15],[24..31].
__device__ __forceinline__ v16h ld_a_frag(const _Float16* buf, int lane, int kbase)
{
    const int m  = lane & 15;
    const int kc = kbase + ((lane >> 4) << 3);
    const _Float16* p = buf + m * 64 + kc;
    v8h lo = *(const v8h*)(p);
    v8h hi = *(const v8h*)(p + 16);
    return __builtin_shufflevector(lo, hi, 0,1,2,3,4,5,6,7,8,9,10,11,12,13,14,15);
}

// B (32x16, f16), weights packed [N][K] in LDS: lane holds column n=lane&15,
// K chunks [(lane>>4)*8 .. +7] and [16+(lane>>4)*8 .. +7] (matches A).
__device__ __forceinline__ v16h ld_b_frag(const _Float16* W, int lane, int ntile,
                                          int K, int kbase)
{
    const _Float16* p = W + (ntile * 16 + (lane & 15)) * K + kbase + ((lane >> 4) << 3);
    v8h lo = *(const v8h*)(p);
    v8h hi = *(const v8h*)(p + 16);
    return __builtin_shufflevector(lo, hi, 0,1,2,3,4,5,6,7,8,9,10,11,12,13,14,15);
}

// One hidden layer, IN-PLACE: D[16x64] = relu(A[16xK] * W[Kx64] + b).
// A-fragments are fully hoisted into VGPRs before any store, so reading and
// writing the same wave-private LDS tile is safe.
template <int K>
__device__ __forceinline__ void gemm_relu(_Float16* __restrict__ buf,
                                          const _Float16* __restrict__ W,
                                          const float* __restrict__ bias, int lane)
{
    const int col = lane & 15;
    const int hs  = lane >> 4;          // C/D: rows hs*8 .. hs*8+7
    v16h a[K / 32];
#pragma unroll
    for (int kb = 0; kb < K; kb += 32)
        a[kb / 32] = ld_a_frag(buf, lane, kb);
#pragma unroll
    for (int nt = 0; nt < 4; ++nt) {
        v8f acc = {};
#pragma unroll
        for (int kb = 0; kb < K; kb += 32) {
            v16h b = ld_b_frag(W, lane, nt, K, kb);
            acc = __builtin_amdgcn_wmma_f32_16x16x32_f16(
                false, a[kb / 32], false, b, (short)0, acc, false, false);
        }
        const int n  = nt * 16 + col;
        const float bv = bias[n];
#pragma unroll
        for (int r = 0; r < 8; ++r) {
            float v = fmaxf(acc[r] + bv, 0.f);
            buf[(hs * 8 + r) * 64 + n] = (_Float16)v;
        }
    }
}

// ---------------------------------------------------------------------------
// Fully fused NGP forward: hash encode + density MLP + color MLP.
// Block = 256 threads = 8 waves; each wave owns a 16-sample tile.
// All packed weights staged once per block into LDS (L2 broadcast cut 8x);
// hash gathers are single-b32 packed-f16 pairs blended with v_pk_fma_f16.
// ---------------------------------------------------------------------------
__global__ __launch_bounds__(256) void ngp_fused(
    const float* __restrict__ x, const _Float16* __restrict__ tabs,
    const _Float16* __restrict__ ws,
    const float* __restrict__ db0, const float* __restrict__ db1,
    const float* __restrict__ db2, const float* __restrict__ cb0,
    const float* __restrict__ cb1, const float* __restrict__ cb2,
    const float* __restrict__ cb3,
    float* __restrict__ outColor, float* __restrict__ outSigma, int Npts)
{
    __shared__ __align__(32) _Float16 wlds[WS_HALVES];      // 36,864 B weights
    __shared__ __align__(32) _Float16 act[8 * 16 * 64];     // 16,384 B acts

    const int tid   = threadIdx.x;
    const int lane  = tid & 31;
    const int wv    = tid >> 5;
    _Float16* buf   = act + wv * 1024;
    const int m     = lane & 15;
    const int sBase = blockIdx.x * 128 + wv * 16;
    const int s     = min(sBase + m, Npts - 1);

    // ---- cooperative weight staging: global (packed) -> LDS, b128 chunks --
    {
        const v8h* src = (const v8h*)ws;
        v8h* dst = (v8h*)wlds;
#pragma unroll
        for (int i = 0; i < WS_HALVES / 8 / 256; ++i)       // 9 iters
            dst[tid + i * 256] = src[tid + i * 256];
    }

    // ---- multires hash encoding (overlaps weight-staging latency) ---------
    const float gx = x[s * 3 + 0] * 0.5f + 0.5f;
    const float gy = x[s * 3 + 1] * 0.5f + 0.5f;
    const float gz = x[s * 3 + 2] * 0.5f + 0.5f;
#pragma unroll
    for (int i = 0; i < 8; ++i) {
        const int l = (lane >> 4) + 2 * i;                 // even/odd level split
        const float res = floorf(32.f * exp2f((float)l * 0.2f));  // 32*8^(l/15)
        const float sc  = res - 1.f;
        const float fx = gx * sc, fy = gy * sc, fz = gz * sc;
        const float cxf = floorf(fx), cyf = floorf(fy), czf = floorf(fz);
        const float wx = fx - cxf, wy = fy - cyf, wz = fz - czf;
        const unsigned cx = (unsigned)(int)cxf;
        const unsigned cy = (unsigned)(int)cyf;
        const unsigned cz = (unsigned)(int)czf;
        const v2h* tl = (const v2h*)(tabs + l * (T_HASH * 2));
        v2h f2 = {};
#pragma unroll
        for (int c = 0; c < 8; ++c) {
            const int di = (c >> 2) & 1, dj = (c >> 1) & 1, dk = c & 1;
            const unsigned h = (cx + (unsigned)di)
                             ^ ((cy + (unsigned)dj) * 2654435761u)
                             ^ ((cz + (unsigned)dk) * 805459861u);
            const unsigned idx = h & (unsigned)(T_HASH - 1);
            const float wt = (di ? wx : 1.f - wx) * (dj ? wy : 1.f - wy)
                           * (dk ? wz : 1.f - wz);
            const _Float16 wh = (_Float16)wt;
            const v2h e = tl[idx];                         // one b32 gather
            f2 += (v2h){wh, wh} * e;                       // v_pk_fma_f16
        }
        *(v2h*)(buf + m * 64 + 2 * l) = f2;                // one b32 LDS store
    }
    __syncthreads();

    // ---- density MLP: 32 -> 64 -> 64 (in-place tiles, weights from LDS) ---
    gemm_relu<32>(buf, wlds + OFF_D0, db0, lane);  __syncthreads();
    gemm_relu<64>(buf, wlds + OFF_D1, db1, lane);  __syncthreads();

    // ---- density head: 64 -> 16 (no relu), sigma = exp(d[:,0]) ------------
    {
        v16h a0 = ld_a_frag(buf, lane, 0);
        v16h a1 = ld_a_frag(buf, lane, 32);
        v8f acc = {};
        acc = __builtin_amdgcn_wmma_f32_16x16x32_f16(
            false, a0, false, ld_b_frag(wlds + OFF_D2, lane, 0, 64, 0),
            (short)0, acc, false, false);
        acc = __builtin_amdgcn_wmma_f32_16x16x32_f16(
            false, a1, false, ld_b_frag(wlds + OFF_D2, lane, 0, 64, 32),
            (short)0, acc, false, false);
        const int n = lane & 15, hs = lane >> 4;
        const float bv = db2[n];
#pragma unroll
        for (int r = 0; r < 8; ++r) {
            const int mm = hs * 8 + r;
            const float v = acc[r] + bv;
            buf[mm * 64 + n]      = (_Float16)v;      // cols 0..15: d
            buf[mm * 64 + 16 + n] = (_Float16)0.f;    // cols 16..31: K-pad
            if (n == 0) outSigma[sBase + mm] = expf(v);
        }
    }
    __syncthreads();

    // ---- color MLP: 16(pad 32) -> 64 -> 64 -> 64 --------------------------
    gemm_relu<32>(buf, wlds + OFF_C0, cb0, lane);  __syncthreads();
    gemm_relu<64>(buf, wlds + OFF_C1, cb1, lane);  __syncthreads();
    gemm_relu<64>(buf, wlds + OFF_C2, cb2, lane);  __syncthreads();

    // ---- color head: 64 -> 3 (N padded to 16), sigmoid --------------------
    {
        v16h a0 = ld_a_frag(buf, lane, 0);
        v16h a1 = ld_a_frag(buf, lane, 32);
        v8f acc = {};
        acc = __builtin_amdgcn_wmma_f32_16x16x32_f16(
            false, a0, false, ld_b_frag(wlds + OFF_C3, lane, 0, 64, 0),
            (short)0, acc, false, false);
        acc = __builtin_amdgcn_wmma_f32_16x16x32_f16(
            false, a1, false, ld_b_frag(wlds + OFF_C3, lane, 0, 64, 32),
            (short)0, acc, false, false);
        const int n = lane & 15, hs = lane >> 4;
        if (n < 3) {
            const float bv = cb3[n];
#pragma unroll
            for (int r = 0; r < 8; ++r) {
                const int mm = hs * 8 + r;
                const float v = acc[r] + bv;
                outColor[(size_t)(sBase + mm) * 3 + n] = 1.f / (1.f + expf(-v));
            }
        }
    }
}

// ---------------------------------------------------------------------------
extern "C" void kernel_launch(void* const* d_in, const int* in_sizes, int n_in,
                              void* d_out, int out_size, void* d_ws, size_t ws_size,
                              hipStream_t stream)
{
    const float* x      = (const float*)d_in[0];
    const float* tables = (const float*)d_in[1];
    const float* dW0 = (const float*)d_in[2];   const float* db0 = (const float*)d_in[3];
    const float* dW1 = (const float*)d_in[4];   const float* db1 = (const float*)d_in[5];
    const float* dW2 = (const float*)d_in[6];   const float* db2 = (const float*)d_in[7];
    const float* cW0 = (const float*)d_in[8];   const float* cb0 = (const float*)d_in[9];
    const float* cW1 = (const float*)d_in[10];  const float* cb1 = (const float*)d_in[11];
    const float* cW2 = (const float*)d_in[12];  const float* cb2 = (const float*)d_in[13];
    const float* cW3 = (const float*)d_in[14];  const float* cb3 = (const float*)d_in[15];

    _Float16* ws   = (_Float16*)d_ws;           // [0, WS_HALVES): weights
    _Float16* tabs = ws + WS_HALVES;            // [WS_HALVES, +TAB_HALVES): tables
    const int Npts = in_sizes[0] / 3;           // N = 1<<20

    pack_weights<<<(WS_HALVES + 255) / 256, 256, 0, stream>>>(
        dW0, dW1, dW2, cW0, cW1, cW2, cW3, ws);
    pack_tables<<<(TAB_HALVES + 255) / 256, 256, 0, stream>>>(tables, tabs);

    float* outColor = (float*)d_out;            // [N,3] then [N,1] sigma
    float* outSigma = outColor + (size_t)Npts * 3;

    const int blocks = (Npts + 127) / 128;      // 128 samples / block (8 waves x 16)
    ngp_fused<<<blocks, 256, 0, stream>>>(
        x, tabs, ws, db0, db1, db2, cb0, cb1, cb2, cb3,
        outColor, outSigma, Npts);
}
